// COM_HGNN_K4_26491358282335
// MI455X (gfx1250) — matrix operand order repaired and checked by reading, not compiled, not run on hardware
//
#include <hip/hip_runtime.h>
#include <hip/hip_bf16.h>

typedef __attribute__((ext_vector_type(16))) _Float16 v16h;
typedef __attribute__((ext_vector_type(8)))  _Float16 v8h;
typedef __attribute__((ext_vector_type(8)))  float    v8f;
typedef __attribute__((ext_vector_type(4)))  unsigned int v4u;

// ---------------------------------------------------------------------------
// Weight prep: build per-layer K-stacked, transposed (N-major) f16 weights.
// Wb: [L][5][128n][128k]  blocks = {Wrel_gt, Wrel_gs, Wrel_gr, Wrel_jb, sum(Wroot)}
// Wj: [L][4][128n][128k]  blocks = {Wrel_bj, Wrel_jj, Wrel_fj, sum(Wroot)}
// Wf: [L][2][128n][128k]  blocks = {Wrel_jf, Wroot_jf}
// ---------------------------------------------------------------------------
__global__ void prep_stack_kernel(const float* __restrict__ Wrel,
                                  const float* __restrict__ Wroot,
                                  _Float16* __restrict__ Wb,
                                  _Float16* __restrict__ Wj,
                                  _Float16* __restrict__ Wf,
                                  int L) {
  int idx = blockIdx.x * blockDim.x + threadIdx.x;
  int total = L * 11 * 16384;
  if (idx >= total) return;
  int l    = idx / (11 * 16384);
  int rem  = idx - l * (11 * 16384);
  int slot = rem >> 14;
  int e    = rem & 16383;
  int n    = e >> 7;
  int k    = e & 127;
  auto rel  = [&](int t) { return Wrel [(((size_t)l * 8 + t) * 128 + k) * 128 + n]; };
  auto root = [&](int t) { return Wroot[(((size_t)l * 8 + t) * 128 + k) * 128 + n]; };
  float v;
  _Float16* dst;
  if (slot < 4) {
    const int bt[4] = {0, 1, 2, 4};
    v = rel(bt[slot]);
    dst = Wb + (size_t)(l * 5 + slot) * 16384;
  } else if (slot == 4) {
    v = root(0) + root(1) + root(2) + root(4);
    dst = Wb + (size_t)(l * 5 + 4) * 16384;
  } else if (slot < 8) {
    const int jt[3] = {3, 5, 7};
    v = rel(jt[slot - 5]);
    dst = Wj + (size_t)(l * 4 + (slot - 5)) * 16384;
  } else if (slot == 8) {
    v = root(3) + root(5) + root(7);
    dst = Wj + (size_t)(l * 4 + 3) * 16384;
  } else if (slot == 9) {
    v = rel(6);
    dst = Wf + (size_t)(l * 2 + 0) * 16384;
  } else {
    v = root(6);
    dst = Wf + (size_t)(l * 2 + 1) * 16384;
  }
  dst[n * 128 + k] = (_Float16)v;
}

__global__ void prep_mlp_kernel(const float* __restrict__ W1,
                                const float* __restrict__ W2,
                                _Float16* __restrict__ W1t,
                                _Float16* __restrict__ W2t) {
  int idx = blockIdx.x * blockDim.x + threadIdx.x;
  if (idx >= 2 * 16384) return;
  int which = idx >> 14;
  int e = idx & 16383;
  int n = e >> 7, k = e & 127;
  const float* s = which ? W2 : W1;
  _Float16*    d = which ? W2t : W1t;
  d[n * 128 + k] = (_Float16)s[k * 128 + n];
}

__global__ void prep_bias_kernel(const float* __restrict__ brel,
                                 float* __restrict__ bb,
                                 float* __restrict__ bj,
                                 float* __restrict__ bf, int L) {
  int idx = blockIdx.x * blockDim.x + threadIdx.x;
  if (idx >= L * 128) return;
  int l = idx >> 7, n = idx & 127;
  auto g = [&](int t) { return brel[((size_t)l * 8 + t) * 128 + n]; };
  bb[idx] = g(0) + g(1) + g(2) + g(4);
  bj[idx] = g(3) + g(5) + g(7);
  bf[idx] = g(6);
}

// ---------------------------------------------------------------------------
// Encoder: out = relu(bias + sym * (x @ W)) in f16
// ---------------------------------------------------------------------------
__global__ void enc_kernel(const float* __restrict__ x, int fin,
                           const float* __restrict__ W,
                           const float* __restrict__ bias,
                           const float* __restrict__ sym,
                           _Float16* __restrict__ out, int n) {
  int id = blockIdx.x * blockDim.x + threadIdx.x;
  if (id >= n * 128) return;
  int row = id >> 7, h = id & 127;
  float d = 0.f;
  for (int k = 0; k < fin; ++k) d += x[(size_t)row * fin + k] * W[k * 128 + h];
  float s = sym ? sym[row % 12] : 1.f;
  float acc = bias[h] + s * d;
  out[id] = (_Float16)fmaxf(acc, 0.f);
}

// ---------------------------------------------------------------------------
// Decoder: out[r,o] = coeff * (dec_b[o] + x_base[r] . dec_W[:,o])
// ---------------------------------------------------------------------------
__global__ void dec_kernel(const _Float16* __restrict__ xb,
                           const float* __restrict__ W,
                           const float* __restrict__ bias,
                           const float* __restrict__ clin,
                           const float* __restrict__ cang,
                           float* __restrict__ out, int nb) {
  int id = blockIdx.x * blockDim.x + threadIdx.x;
  if (id >= nb * 6) return;
  int row = id / 6, o = id - row * 6;
  float d = bias[o];
  for (int k = 0; k < 128; ++k) d += (float)xb[(size_t)row * 128 + k] * W[k * 6 + o];
  int i = row & 3;
  float c = (o < 3) ? clin[i * 3 + o] : cang[i * 3 + (o - 3)];
  out[id] = d * c;
}

// ---------------------------------------------------------------------------
// Gather: compute A-row pointer(s) for block b, row r.
// GM: 0=base (5 blocks), 1=joint (4 blocks), 2=foot (2 blocks), 3=direct
// ---------------------------------------------------------------------------
template <int GM>
__device__ __forceinline__ void gather_row(int b, int r, int g12, int leg, int pos,
                                           const _Float16* xb, const _Float16* xj,
                                           const _Float16* xf, const _Float16* Adir,
                                           const _Float16*& a1p, const _Float16*& a2p) {
  a1p = nullptr;
  a2p = nullptr;
  if constexpr (GM == 0) {  // base: gt/gs/gr XOR perms, jb = joint 3r, summed root
    if (b == 0)      a1p = xb + (size_t)(r ^ 1) * 128;
    else if (b == 1) a1p = xb + (size_t)(r ^ 2) * 128;
    else if (b == 2) a1p = xb + (size_t)(r ^ 3) * 128;
    else if (b == 3) a1p = xj + (size_t)(3 * r) * 128;
    else             a1p = xb + (size_t)r * 128;
  } else if constexpr (GM == 1) {  // joint: bj / jj-chain / fj / root
    if (b == 0) {
      if (pos == 0) a1p = xb + (size_t)(4 * g12 + leg) * 128;
    } else if (b == 1) {
      if (pos != 2) a1p = xj + (size_t)(r + 1) * 128;
      if (pos != 0) a2p = xj + (size_t)(r - 1) * 128;
    } else if (b == 2) {
      if (pos == 2) a1p = xf + (size_t)(4 * g12 + leg) * 128;
    } else {
      a1p = xj + (size_t)r * 128;
    }
  } else if constexpr (GM == 2) {  // foot: jf = joint 3r+2, root
    a1p = (b == 0) ? (xj + (size_t)(3 * r + 2) * 128) : (xf + (size_t)r * 128);
  } else {  // direct A
    a1p = Adir + (size_t)r * 128;
  }
}

__device__ __forceinline__ v16h load_afrag(const _Float16* a1p, const _Float16* a2p,
                                           int kloc) {
  const v8h hz = {(_Float16)0, (_Float16)0, (_Float16)0, (_Float16)0,
                  (_Float16)0, (_Float16)0, (_Float16)0, (_Float16)0};
  v8h a0 = hz, a1 = hz;
  if (a1p) {
    a0 = *(const v8h*)(a1p + kloc);
    a1 = *(const v8h*)(a1p + kloc + 16);
  }
  if (a2p) {
    a0 = a0 + *(const v8h*)(a2p + kloc);
    a1 = a1 + *(const v8h*)(a2p + kloc + 16);
  }
  v16h A;
#pragma unroll
  for (int i = 0; i < 8; ++i) { A[i] = a0[i]; A[i + 8] = a1[i]; }
  return A;
}

// ---------------------------------------------------------------------------
// Fused gathered-A GEMM + bias/ReLU/residual epilogue (f16 WMMA, f32 accum).
// EPI: bit0 = relu, bit1 = add residual (f16)
// W layout: [NBLK][128 n][128 k] f16 (N-major per block).
// 8 waves / workgroup; each wave owns a 32-row x 128-col tile (16 accumulators).
// B tile (32K x 128N) staged in LDS via global_load_async_to_lds (ASYNCcnt).
// ---------------------------------------------------------------------------
template <int GM, int NBLK, int EPI>
__global__ __launch_bounds__(256) void hgnn_gemm_kernel(
    const _Float16* __restrict__ xb, const _Float16* __restrict__ xj,
    const _Float16* __restrict__ xf, const _Float16* __restrict__ Adir,
    const _Float16* __restrict__ W, const float* __restrict__ bias,
    const _Float16* __restrict__ resid, _Float16* __restrict__ out) {
  __shared__ _Float16 ldsB[128 * 32];  // 8 KB: one 32K x 128N weight tile

  const int lane = threadIdx.x & 31;
  const int wave = threadIdx.x >> 5;
  const int m    = lane & 15;
  const int hi   = lane >> 4;
  const int r0   = (blockIdx.x * 8 + wave) * 32;
  const int rA   = r0 + m;       // rows of first 16-row tile
  const int rB   = r0 + 16 + m;  // rows of second 16-row tile

  // joint-node decomposition (used when GM==1)
  const int gA = rA / 12, pA = rA - gA * 12, legA = pA / 3, posA = pA - legA * 3;
  const int gB = rB / 12, pB = rB - gB * 12, legB = pB / 3, posB = pB - legB * 3;

  const v8f vzero = {0.f, 0.f, 0.f, 0.f, 0.f, 0.f, 0.f, 0.f};
  v8f acc0[8], acc1[8];
#pragma unroll
  for (int i = 0; i < 8; ++i) { acc0[i] = vzero; acc1[i] = vzero; }

  // per-thread async-fill addressing (thread t copies 32B of the B tile)
  const int fn    = threadIdx.x >> 1;        // B row n = 0..127
  const int fpart = threadIdx.x & 1;         // 16-half chunk
  const unsigned ldsoff =
      (unsigned)(uintptr_t)&ldsB[fn * 32 + fpart * 16];

  const int KT = NBLK * 4;  // 32-wide K tiles
  for (int kt = 0; kt < KT; ++kt) {
    __syncthreads();  // previous tile's LDS reads complete before overwrite
    {
      const _Float16* src = W + (size_t)(kt >> 2) * 16384 + (size_t)fn * 128 +
                            (kt & 3) * 32 + fpart * 16;
      // CDNA5 async global->LDS copy (32 contiguous bytes per thread);
      // inst offset applies to both global and LDS addresses.
      asm volatile(
          "global_load_async_to_lds_b128 %0, %1, off\n\t"
          "global_load_async_to_lds_b128 %0, %1, off offset:16\n\t"
          "s_wait_asynccnt 0x0"
          :
          : "v"(ldsoff), "v"(src)
          : "memory");
    }
    __syncthreads();

    const int b    = kt >> 2;
    const int kloc = (kt & 3) * 32 + hi * 8;

    const _Float16 *p1a, *p2a, *p1b, *p2b;
    gather_row<GM>(b, rA, gA, legA, posA, xb, xj, xf, Adir, p1a, p2a);
    gather_row<GM>(b, rB, gB, legB, posB, xb, xj, xf, Adir, p1b, p2b);
    v16h A0 = load_afrag(p1a, p2a, kloc);
    v16h A1 = load_afrag(p1b, p2b, kloc);

    // preload all 8 B fragments into distinct registers, then issue WMMAs
    v16h Bf[8];
#pragma unroll
    for (int nt = 0; nt < 8; ++nt) {
      const _Float16* bp = &ldsB[(size_t)(nt * 16 + m) * 32 + hi * 8];
      v8h b0 = *(const v8h*)bp;
      v8h b1 = *(const v8h*)(bp + 16);
#pragma unroll
      for (int i = 0; i < 8; ++i) { Bf[nt][i] = b0[i]; Bf[nt][i + 8] = b1[i]; }
    }
#pragma unroll
    for (int nt = 0; nt < 8; ++nt) {
      acc0[nt] = __builtin_amdgcn_wmma_f32_16x16x32_f16(
          false, A0, false, Bf[nt], (short)0, acc0[nt], false, false);
      acc1[nt] = __builtin_amdgcn_wmma_f32_16x16x32_f16(
          false, A1, false, Bf[nt], (short)0, acc1[nt], false, false);
    }
  }

  // epilogue: bias (+relu) (+residual), store f16
#pragma unroll
  for (int nt = 0; nt < 8; ++nt) {
    const int n = nt * 16 + m;
    const float bn = bias[n];
#pragma unroll
    for (int i = 0; i < 8; ++i) {
      {
        int row = r0 + i + hi * 8;
        float v = acc0[nt][i] + bn;
        if constexpr (EPI & 1) v = fmaxf(v, 0.f);
        if constexpr (EPI & 2) v += (float)resid[(size_t)row * 128 + n];
        out[(size_t)row * 128 + n] = (_Float16)v;
      }
      {
        int row = r0 + 16 + i + hi * 8;
        float v = acc1[nt][i] + bn;
        if constexpr (EPI & 1) v = fmaxf(v, 0.f);
        if constexpr (EPI & 2) v += (float)resid[(size_t)row * 128 + n];
        out[(size_t)row * 128 + n] = (_Float16)v;
      }
    }
  }
}

// ---------------------------------------------------------------------------
extern "C" void kernel_launch(void* const* d_in, const int* in_sizes, int n_in,
                              void* d_out, int out_size, void* d_ws, size_t ws_size,
                              hipStream_t stream) {
  const float* x_base      = (const float*)d_in[0];
  const float* x_joint     = (const float*)d_in[1];
  const float* x_foot      = (const float*)d_in[2];
  const float* joint_sym_w = (const float*)d_in[3];
  const float* coeff_lin   = (const float*)d_in[4];
  const float* coeff_ang   = (const float*)d_in[5];
  const float* enc_W_base  = (const float*)d_in[6];
  const float* enc_b_base  = (const float*)d_in[7];
  const float* enc_W_joint = (const float*)d_in[8];
  const float* enc_b_joint = (const float*)d_in[9];
  const float* enc_W_foot  = (const float*)d_in[10];
  const float* enc_b_foot  = (const float*)d_in[11];
  const float* conv_W_rel  = (const float*)d_in[12];
  const float* conv_b_rel  = (const float*)d_in[13];
  const float* conv_W_root = (const float*)d_in[14];
  const float* bt_W1       = (const float*)d_in[15];
  const float* bt_b1       = (const float*)d_in[16];
  const float* bt_W2       = (const float*)d_in[17];
  const float* bt_b2       = (const float*)d_in[18];
  const float* dec_W       = (const float*)d_in[19];
  const float* dec_b       = (const float*)d_in[20];
  // edge indices d_in[21..28] are fixed K4 structure -> computed analytically

  const int NB = in_sizes[0] / 6;
  const int NJ = in_sizes[1] / 2;
  const int NF = in_sizes[2] / 2;
  const int L  = 4;

  char* ws = (char*)d_ws;
  size_t off = 0;
  auto alloc = [&](size_t bytes) -> char* {
    off = (off + 255) & ~(size_t)255;
    char* p = ws + off;
    off += bytes;
    return p;
  };

  _Float16* xb16[2] = {(_Float16*)alloc((size_t)NB * 128 * 2),
                       (_Float16*)alloc((size_t)NB * 128 * 2)};
  _Float16* xj16[2] = {(_Float16*)alloc((size_t)NJ * 128 * 2),
                       (_Float16*)alloc((size_t)NJ * 128 * 2)};
  _Float16* xf16[2] = {(_Float16*)alloc((size_t)NF * 128 * 2),
                       (_Float16*)alloc((size_t)NF * 128 * 2)};
  _Float16* nb16 = (_Float16*)alloc((size_t)NB * 128 * 2);
  _Float16* h16  = (_Float16*)alloc((size_t)NB * 128 * 2);
  _Float16* Wb  = (_Float16*)alloc((size_t)L * 5 * 16384 * 2);
  _Float16* Wj  = (_Float16*)alloc((size_t)L * 4 * 16384 * 2);
  _Float16* Wf  = (_Float16*)alloc((size_t)L * 2 * 16384 * 2);
  _Float16* W1t = (_Float16*)alloc(16384 * 2);
  _Float16* W2t = (_Float16*)alloc(16384 * 2);
  float* bb = (float*)alloc(L * 128 * 4);
  float* bj = (float*)alloc(L * 128 * 4);
  float* bf = (float*)alloc(L * 128 * 4);

  // weight / bias prep
  {
    int tot = L * 11 * 16384;
    prep_stack_kernel<<<(tot + 255) / 256, 256, 0, stream>>>(
        conv_W_rel, conv_W_root, Wb, Wj, Wf, L);
    prep_mlp_kernel<<<(2 * 16384 + 255) / 256, 256, 0, stream>>>(bt_W1, bt_W2, W1t, W2t);
    prep_bias_kernel<<<(L * 128 + 255) / 256, 256, 0, stream>>>(conv_b_rel, bb, bj, bf, L);
  }

  // encoders
  enc_kernel<<<((size_t)NB * 128 + 255) / 256, 256, 0, stream>>>(
      x_base, 6, enc_W_base, enc_b_base, nullptr, xb16[0], NB);
  enc_kernel<<<((size_t)NJ * 128 + 255) / 256, 256, 0, stream>>>(
      x_joint, 2, enc_W_joint, enc_b_joint, joint_sym_w, xj16[0], NJ);
  enc_kernel<<<((size_t)NF * 128 + 255) / 256, 256, 0, stream>>>(
      x_foot, 2, enc_W_foot, enc_b_foot, nullptr, xf16[0], NF);

  int cur = 0;
  for (int l = 0; l < L; ++l) {
    int nxt = cur ^ 1;
    // joint: relu(new)+x   (K=512)
    hgnn_gemm_kernel<1, 4, 3><<<NJ / 256, 256, 0, stream>>>(
        xb16[cur], xj16[cur], xf16[cur], nullptr,
        Wj + (size_t)l * 4 * 16384, bj + l * 128, xj16[cur], xj16[nxt]);
    // foot: relu(new)+x    (K=256)
    hgnn_gemm_kernel<2, 2, 3><<<NF / 256, 256, 0, stream>>>(
        xb16[cur], xj16[cur], xf16[cur], nullptr,
        Wf + (size_t)l * 2 * 16384, bf + l * 128, xf16[cur], xf16[nxt]);
    // base pre-activation  (K=640)
    hgnn_gemm_kernel<0, 5, 0><<<NB / 256, 256, 0, stream>>>(
        xb16[cur], xj16[cur], xf16[cur], nullptr,
        Wb + (size_t)l * 5 * 16384, bb + l * 128, nullptr, nb16);
    // base MLP layer 1: relu(nb @ W1 + b1)
    hgnn_gemm_kernel<3, 1, 1><<<NB / 256, 256, 0, stream>>>(
        nullptr, nullptr, nullptr, nb16, W1t, bt_b1, nullptr, h16);
    // base MLP layer 2 + residual: h @ W2 + b2 + x_base
    hgnn_gemm_kernel<3, 1, 2><<<NB / 256, 256, 0, stream>>>(
        nullptr, nullptr, nullptr, h16, W2t, bt_b2, xb16[cur], xb16[nxt]);
    cur = nxt;
  }

  // decoder + symmetry coefficients -> f32 output [B,4,6]
  dec_kernel<<<((size_t)NB * 6 + 255) / 256, 256, 0, stream>>>(
      xb16[cur], dec_W, dec_b, coeff_lin, coeff_ang, (float*)d_out, NB);
}
